// QuantumFeedForward_65481071410396
// MI455X (gfx1250) — matrix operand order repaired and checked by reading, not compiled, not run on hardware
//
#include <hip/hip_runtime.h>

// ---------------------------------------------------------------------------
// QuantumFeedForward for MI455X (gfx1250, wave32, WMMA, async LDS staging).
//
// Math reduction: RX layer -> product state; CNOT ring -> GF(2) linear basis
// permutation, so Z expvals collapse to:
//     c_w = cos(theta_w);  z_0 = c1*...*c7;  z_q = c0*...*c_q (q>=1)
//
// Pipeline (all on `stream`):
//   1) qffn_encode_z : theta = x @ Wenc^T (HBM-bound, 134MB) -> z [T,8] f32
//   2) qffn_pack_wdec: W_dec -> f16 row-pairs (half2 per q) for v_pk_fma_f16
//   3) qffn_pack_wout: W_out -> f16, pre-swizzled into WMMA B-fragment order
//   4) qffn_fused_gemm: out = relu(z @ Wdec^T) @ Wout^T
//        - h built directly in A-fragment registers (pk_fma/pk_max, co-exec)
//        - GEMM3 via v_wmma_f32_16x16x32_f16 (f32 accumulate)
//        - B + Wdec staged global->LDS with GLOBAL_LOAD_ASYNC_TO_LDS_B128,
//          double buffered, one barrier per K-step
// ---------------------------------------------------------------------------

#define BB 8
#define SS 4096
#define TT (BB * SS)      // 32768 tokens
#define EMBED 1024
#define FFN 4096
#define NQ 8

typedef _Float16 h2   __attribute__((ext_vector_type(2)));
typedef _Float16 v16h __attribute__((ext_vector_type(16)));
typedef float    v8f  __attribute__((ext_vector_type(8)));

static __device__ __forceinline__ h2 h2splat(_Float16 v) { h2 r; r.x = v; r.y = v; return r; }

// Generic shared pointer -> 32-bit LDS offset (the low 32 bits of the flat address).
static __device__ __forceinline__ unsigned lds_off(const void* p) {
    return (unsigned)(size_t)p;
}
// CDNA5 async DMA: memory -> LDS, tracked by ASYNCcnt (no VGPR round-trip).
static __device__ __forceinline__ void async_g2l_b128(unsigned lds, const void* g) {
    asm volatile("global_load_async_to_lds_b128 %0, %1, off" :: "v"(lds), "v"(g) : "memory");
}
static __device__ __forceinline__ void wait_async0() {
    asm volatile("s_wait_asynccnt 0" ::: "memory");
}

// ---------------------------------------------------------------------------
// Kernel 1: theta = x @ Wenc^T, then z via cos + prefix products.
// One wave per token; x read fully coalesced (float4 per lane).
// ---------------------------------------------------------------------------
__global__ __launch_bounds__(256)
void qffn_encode_z(const float* __restrict__ x,
                   const float* __restrict__ Wenc,
                   float* __restrict__ z)
{
    __shared__ float4 wsh[NQ * EMBED / 4];   // 32KB
    const int tid = threadIdx.x;
    const float4* w4 = (const float4*)Wenc;
    for (int i = tid; i < NQ * EMBED / 4; i += 256) wsh[i] = w4[i];
    __syncthreads();

    const int wave = tid >> 5, lane = tid & 31;
    const int t = blockIdx.x * 8 + wave;
    const float4* x4 = (const float4*)(x + (size_t)t * EMBED);

    float acc[NQ];
    #pragma unroll
    for (int q = 0; q < NQ; q++) acc[q] = 0.f;

    #pragma unroll
    for (int i = 0; i < EMBED / 128; i++) {        // 8 iterations
        float4 xv = x4[lane + i * 32];
        #pragma unroll
        for (int q = 0; q < NQ; q++) {
            float4 wv = wsh[q * (EMBED / 4) + lane + i * 32];
            acc[q] += xv.x * wv.x + xv.y * wv.y + xv.z * wv.z + xv.w * wv.w;
        }
    }
    #pragma unroll
    for (int q = 0; q < NQ; q++) {
        #pragma unroll
        for (int off = 16; off > 0; off >>= 1) acc[q] += __shfl_xor(acc[q], off, 32);
    }

    if (lane == 0) {
        float c[NQ], zo[NQ];
        #pragma unroll
        for (int q = 0; q < NQ; q++) c[q] = __cosf(acc[q]);
        float p = c[0];
        #pragma unroll
        for (int q = 1; q < NQ; q++) { p *= c[q]; zo[q] = p; }
        float s = 1.f;
        #pragma unroll
        for (int q = 1; q < NQ; q++) s *= c[q];
        zo[0] = s;
        #pragma unroll
        for (int q = 0; q < NQ; q++) z[(size_t)t * NQ + q] = zo[q];
    }
}

// ---------------------------------------------------------------------------
// Kernel 2: pack W_dec[4096,8] f32 -> wd[2048 pairs][8 q] as half2:
//   wd[fp][q] = { Wdec[2fp][q], Wdec[2fp+1][q] }  (feeds v_pk_fma_f16)
// ---------------------------------------------------------------------------
__global__ __launch_bounds__(256)
void qffn_pack_wdec(const float* __restrict__ Wdec, h2* __restrict__ wd)
{
    const int i = blockIdx.x * 256 + threadIdx.x;   // 16384 total
    const int fp = i >> 3, q = i & 7;
    h2 v;
    v.x = (_Float16)Wdec[(size_t)(2 * fp) * NQ + q];
    v.y = (_Float16)Wdec[(size_t)(2 * fp + 1) * NQ + q];
    wd[i] = v;
}

// ---------------------------------------------------------------------------
// Kernel 3: pack W_out[1024,4096] f32 -> f16 pre-swizzled B fragments.
// Chunk (n16, kc) = one 32x16 (KxN) B tile, 512 halfs; lane L reads its whole
// v16h fragment as 32 contiguous bytes at chunk + L*32B.
// B layout (contiguous k, same order as the A build): k = 16*(L/16) + j2, n = L%16.
// ---------------------------------------------------------------------------
__global__ __launch_bounds__(256)
void qffn_pack_wout(const float* __restrict__ Wout, _Float16* __restrict__ bp)
{
    const int i = blockIdx.x * 256 + threadIdx.x;   // EMBED*FFN = 4,194,304
    const int chunk = i >> 9;          // (n16*128 + kc), 8192 chunks
    const int r = i & 511;
    const int L = r >> 4, j2 = r & 15;
    const int n16 = chunk >> 7, kc = chunk & 127;
    const int khalf = L >> 4, nl = L & 15;
    const int e = n16 * 16 + nl;
    const int k = kc * 32 + khalf * 16 + j2;
    bp[i] = (_Float16)Wout[(size_t)e * FFN + k];
}

// ---------------------------------------------------------------------------
// Kernel 4: fused  out = relu(z @ Wdec^T) @ Wout^T
// Grid (128, 8); block 256 = 8 waves. WG tile = 256 tokens x 128 cols.
// Each wave: 32 tokens x 128 cols = 2 M-frags x 8 N-frags (128 acc VGPRs).
// Per K-step (32): async-stage 8KB B + 512B wd (double buffered), build 2
// A-frags with pk_fma/pk_max, issue 16 v_wmma_f32_16x16x32_f16.
// ---------------------------------------------------------------------------
__global__ __launch_bounds__(256)
void qffn_fused_gemm(const float* __restrict__ z,
                     const h2* __restrict__ wd,
                     const _Float16* __restrict__ bp,
                     float* __restrict__ out)
{
    __shared__ v16h bsv[2][8 * 32];              // 2 x 8KB: 8 B-fragments each
    __shared__ __align__(16) h2 wds[2][16 * NQ]; // 2 x 512B: W_dec pairs per K-step

    const int tid  = threadIdx.x;
    const int wave = tid >> 5, lane = tid & 31;
    const int khalf = lane >> 4, ml = lane & 15;
    const int wgM = blockIdx.x;      // 0..127 -> tokens
    const int wgN = blockIdx.y;      // 0..7   -> cols (128 each)

    // Per-lane z splats for the two M-fragments (row = lane%16 of each frag).
    h2 z2[2][NQ];
    #pragma unroll
    for (int mf = 0; mf < 2; mf++) {
        const int t = wgM * 256 + wave * 32 + mf * 16 + ml;
        const float* zr = z + (size_t)t * NQ;
        #pragma unroll
        for (int q = 0; q < NQ; q++) z2[mf][q] = h2splat((_Float16)zr[q]);
    }

    v8f acc[2][8];
    const v8f zero8 = {};
    #pragma unroll
    for (int mf = 0; mf < 2; mf++)
        #pragma unroll
        for (int nt = 0; nt < 8; nt++) acc[mf][nt] = zero8;

    // ---- async staging setup -------------------------------------------------
    const uint4* bp4 = (const uint4*)bp;   // 64 uint4 per 512-half chunk
    const uint4* wd4 = (const uint4*)wd;   // 32 uint4 per K-step (128 h2)

    // Each thread moves 2x16B of B per K-step (8KB total) + threads 0..31 move wd.
    const int c0 = tid >> 6;               // chunk 0..3
    const uint4* bsrc0 = bp4 + ((size_t)(wgN * 8 + c0)     * 128) * 64 + (tid & 63);
    const uint4* bsrc1 = bp4 + ((size_t)(wgN * 8 + c0 + 4) * 128) * 64 + (tid & 63);
    unsigned bdst0[2], bdst1[2], wdst[2];
    #pragma unroll
    for (int b = 0; b < 2; b++) {
        bdst0[b] = lds_off(&((uint4*)&bsv[b][0])[tid]);
        bdst1[b] = bdst0[b] + 256 * sizeof(uint4);
        wdst[b]  = lds_off(&((uint4*)&wds[b][0])[tid & 31]);
    }
    const uint4* wsrc = wd4 + (tid & 31);

    #define STAGE(kc_, buf_)                                                     \
        do {                                                                     \
            async_g2l_b128(bdst0[buf_], bsrc0 + (size_t)(kc_) * 64);             \
            async_g2l_b128(bdst1[buf_], bsrc1 + (size_t)(kc_) * 64);             \
            if (tid < 32) async_g2l_b128(wdst[buf_], wsrc + (size_t)(kc_) * 32); \
        } while (0)

    STAGE(0, 0);
    wait_async0();
    __syncthreads();

    for (int kc = 0; kc < FFN / 32; kc++) {
        const int cur = kc & 1;
        if (kc + 1 < FFN / 32) STAGE(kc + 1, cur ^ 1);   // prefetch next K-step

        // Build A fragments: h[t,f] = relu(sum_q z[t,q]*Wdec[f,q]) in f16 pairs.
        // A-layout (contiguous, matches B pack): VGPR j half h -> k = 16*khalf + 2*j + h
        v16h a[2];
        #pragma unroll
        for (int mf = 0; mf < 2; mf++) {
            #pragma unroll
            for (int j = 0; j < 8; j++) {
                const int p = khalf * 8 + j;  // pair idx 0..15
                h2 s = h2splat((_Float16)0);
                #pragma unroll
                for (int q = 0; q < NQ; q++)
                    s = __builtin_elementwise_fma(wds[cur][p * NQ + q], z2[mf][q], s);
                h2 zz = h2splat((_Float16)0);
                h2 r = __builtin_elementwise_max(s, zz);   // ReLU (v_pk_max)
                a[mf][2 * j]     = r.x;
                a[mf][2 * j + 1] = r.y;
            }
        }

        #pragma unroll
        for (int nt = 0; nt < 8; nt++) {
            const v16h b = bsv[cur][nt * 32 + lane];
            #pragma unroll
            for (int mf = 0; mf < 2; mf++) {
                acc[mf][nt] = __builtin_amdgcn_wmma_f32_16x16x32_f16(
                    false, a[mf], false, b, (short)0, acc[mf][nt], false, false);
            }
        }

        wait_async0();      // prefetch for kc+1 landed (overlapped with compute)
        __syncthreads();    // all waves done reading buffer `cur`
    }
    #undef STAGE

    // Epilogue: D layout -> row = r + 8*khalf, col = lane%16
    #pragma unroll
    for (int mf = 0; mf < 2; mf++) {
        #pragma unroll
        for (int nt = 0; nt < 8; nt++) {
            #pragma unroll
            for (int r = 0; r < 8; r++) {
                const int row = wgM * 256 + wave * 32 + mf * 16 + khalf * 8 + r;
                const int col = wgN * 128 + nt * 16 + ml;
                out[(size_t)row * EMBED + col] = acc[mf][nt][r];
            }
        }
    }
}

// ---------------------------------------------------------------------------
extern "C" void kernel_launch(void* const* d_in, const int* in_sizes, int n_in,
                              void* d_out, int out_size, void* d_ws, size_t ws_size,
                              hipStream_t stream)
{
    (void)in_sizes; (void)n_in; (void)out_size; (void)ws_size;
    const float* x    = (const float*)d_in[0];   // [8,4096,1024]
    const float* Wenc = (const float*)d_in[1];   // [8,1024]
    const float* Wdec = (const float*)d_in[2];   // [4096,8]
    const float* Wout = (const float*)d_in[3];   // [1024,4096]
    float* out = (float*)d_out;                  // [8,4096,1024]

    char* ws = (char*)d_ws;
    float*    zbuf = (float*)ws;                                             // 1MB
    h2*       wd   = (h2*)(ws + (size_t)TT * NQ * sizeof(float));            // 64KB
    _Float16* bpk  = (_Float16*)(ws + (size_t)TT * NQ * sizeof(float)
                                    + (size_t)(FFN / 2) * NQ * sizeof(h2));  // 8MB

    qffn_encode_z<<<TT / 8, 256, 0, stream>>>(x, Wenc, zbuf);
    qffn_pack_wdec<<<(FFN / 2) * NQ / 256, 256, 0, stream>>>(Wdec, wd);
    qffn_pack_wout<<<(EMBED * FFN) / 256, 256, 0, stream>>>(Wout, bpk);

    dim3 grid(TT / 256, EMBED / 128);
    qffn_fused_gemm<<<grid, 256, 0, stream>>>(zbuf, wd, bpk, out);
}
